// Encoder_72919954751934
// MI455X (gfx1250) — matrix operand order repaired and checked by reading, not compiled, not run on hardware
//
#include <hip/hip_runtime.h>
#include <hip/hip_bf16.h>
#include <math.h>

// ---------------------------------------------------------------- types ----
typedef __attribute__((ext_vector_type(16))) __bf16 v16bf;
typedef __attribute__((ext_vector_type(8)))  float  v8f;

union ABF { v16bf v; unsigned int u[8]; };

__device__ __forceinline__ unsigned short f2bf(float f) {
    unsigned int x = __float_as_uint(f);
    x += 0x7FFFu + ((x >> 16) & 1u);          // round-to-nearest-even
    return (unsigned short)(x >> 16);
}
__device__ __forceinline__ float sigm(float x) { return 1.0f / (1.0f + __expf(-x)); }

// --------------------------------------------------- embedding gather -----
// x: [T][B][E] bf16 ; idx: [B][T] ; emb: [V][E] f32
__global__ void k_embed_gather(const int* __restrict__ idx,
                               const float* __restrict__ emb,
                               unsigned short* __restrict__ x) {
    int bid = blockIdx.x;             // t*B + b
    int t = bid >> 6, b = bid & 63;
    int id = idx[b * 512 + t];
    const float* src = emb + (size_t)id * 512;
    unsigned short* dst = x + (size_t)bid * 512;
    for (int e = threadIdx.x; e < 512; e += blockDim.x)
        dst[e] = f2bf(src[e]);
}

// ------------------------------------------- weight transpose + convert ---
// in: [K][N] f32 row-major  ->  out: [N][K] bf16 row-major
__global__ void k_transpose_bf16(const float* __restrict__ in,
                                 unsigned short* __restrict__ out,
                                 int K, int N) {
    int n = blockIdx.x;
    for (int k = threadIdx.x; k < K; k += blockDim.x)
        out[(size_t)n * K + k] = f2bf(in[(size_t)k * N + n]);
}

// ---------------------------------------------------------------- zero ----
__global__ void k_zero(unsigned int* __restrict__ p, int n) {
    int i = blockIdx.x * blockDim.x + threadIdx.x;
    if (i < n) p[i] = 0u;
}

// ---------------------------------------------- input-projection GEMM -----
// C[M][N] f32 = A[M][K] bf16  x  Bt[N][K] bf16 (B stored transposed)
// block = 8 waves; wave w -> rows blockIdx.y*128 + w*16, cols blockIdx.x*64
__global__ __launch_bounds__(256)
void k_gemm_xz(const unsigned short* __restrict__ A,
               const unsigned short* __restrict__ Bt,
               float* __restrict__ C,
               int M, int N, int K) {
    const int lane = threadIdx.x & 31;
    const int wv   = threadIdx.x >> 5;
    const int n16  = lane & 15;
    const int half = lane >> 4;
    const int row0 = blockIdx.y * 128 + wv * 16;
    const int col0 = blockIdx.x * 64;

    v8f acc[4] = {};

    const unsigned short* arow = A + (size_t)(row0 + n16) * K + half * 8;
    for (int k0 = 0; k0 < K; k0 += 32) {
        ABF av;
        *(uint4*)&av.u[0] = *(const uint4*)(arow + k0);        // k = half*8 + 0..7
        *(uint4*)&av.u[4] = *(const uint4*)(arow + k0 + 16);   // k = 16 + half*8 + 0..7
#pragma unroll
        for (int a = 0; a < 4; ++a) {
            const unsigned short* brow =
                Bt + (size_t)(col0 + a * 16 + n16) * K + half * 8 + k0;
            ABF bv;
            *(uint4*)&bv.u[0] = *(const uint4*)(brow);         // k = half*8 + 0..7
            *(uint4*)&bv.u[4] = *(const uint4*)(brow + 16);    // k = 16 + half*8 + 0..7
            acc[a] = __builtin_amdgcn_wmma_f32_16x16x32_bf16(
                false, av.v, false, bv.v, (short)0, acc[a], false, false);
        }
    }
#pragma unroll
    for (int a = 0; a < 4; ++a)
#pragma unroll
        for (int j = 0; j < 8; ++j)
            C[(size_t)(row0 + half * 8 + j) * N + col0 + a * 16 + n16] = acc[a][j];
}

// -------------------------------------------------------- LSTM step -------
// One launch per timestep; fuses forward+backward direction of one layer.
// grid.x = 64 : d = blk>>5 (0=f,1=b), j0 = (blk&31)*16 h-columns owned.
// 8 waves: wave w -> gate g=w>>1, row half m0=(w&1)*32 ; 2 acc tiles, K=512.
__global__ __launch_bounds__(256)
void k_lstm_step(const float* __restrict__ xz_f, const float* __restrict__ xz_b,
                 const unsigned short* __restrict__ Ut_f, const unsigned short* __restrict__ Ut_b,
                 const float* __restrict__ bias_f, const float* __restrict__ bias_b,
                 const unsigned short* __restrict__ hprev_f, const unsigned short* __restrict__ hprev_b,
                 unsigned short* __restrict__ hnext_f, unsigned short* __restrict__ hnext_b,
                 float* __restrict__ c_f, float* __restrict__ c_b,
                 unsigned short* __restrict__ l1out,   // layer1: [T][B][2H] bf16, else null
                 float* __restrict__ encout,           // layer2: [B][T][2H] f32, else null
                 float* __restrict__ stateout,         // layer2 last step: h then c, else null
                 int s) {
    const int H = 512, T = 512;
    const int d  = blockIdx.x >> 5;
    const int j0 = (blockIdx.x & 31) << 4;
    const float*          xz    = d ? xz_b   : xz_f;
    const unsigned short* Ut    = d ? Ut_b   : Ut_f;
    const float*          bias  = d ? bias_b : bias_f;
    const unsigned short* hprev = d ? hprev_b : hprev_f;
    unsigned short*       hnext = d ? hnext_b : hnext_f;
    float*                cst   = d ? c_b    : c_f;
    const int tf = d ? (T - 1 - s) : s;   // time index this scan step maps to

    __shared__ float z_lds[64][68];

    const int tid  = threadIdx.x;
    const int lane = tid & 31, wv = tid >> 5;
    const int g = wv >> 1, m0 = (wv & 1) * 32;
    const int n16 = lane & 15, half = lane >> 4;

    v8f acc0 = {}, acc1 = {};
    const unsigned short* arow0 = hprev + (size_t)(m0 + n16) * H + half * 8;
    const unsigned short* arow1 = arow0 + 16 * H;
    const unsigned short* brow  = Ut + (size_t)(g * H + j0 + n16) * H + half * 8;
    for (int k0 = 0; k0 < H; k0 += 32) {
        ABF av0, av1, bv;
        *(uint4*)&bv.u[0]  = *(const uint4*)(brow + k0);
        *(uint4*)&bv.u[4]  = *(const uint4*)(brow + k0 + 16);
        *(uint4*)&av0.u[0] = *(const uint4*)(arow0 + k0);
        *(uint4*)&av0.u[4] = *(const uint4*)(arow0 + k0 + 16);
        *(uint4*)&av1.u[0] = *(const uint4*)(arow1 + k0);
        *(uint4*)&av1.u[4] = *(const uint4*)(arow1 + k0 + 16);
        acc0 = __builtin_amdgcn_wmma_f32_16x16x32_bf16(
            false, av0.v, false, bv.v, (short)0, acc0, false, false);
        acc1 = __builtin_amdgcn_wmma_f32_16x16x32_bf16(
            false, av1.v, false, bv.v, (short)0, acc1, false, false);
    }
#pragma unroll
    for (int j = 0; j < 8; ++j) {
        z_lds[m0 + half * 8 + j][g * 16 + n16]      = acc0[j];
        z_lds[m0 + 16 + half * 8 + j][g * 16 + n16] = acc1[j];
    }
    __syncthreads();

    for (int it = tid; it < 64 * 16; it += 256) {
        const int m = it >> 4, jj = it & 15;
        const int col = j0 + jj;
        const float* xzr = xz + ((size_t)tf * 64 + m) * 2048;
        float zi = z_lds[m][jj]      + xzr[col]         + bias[col];
        float zf = z_lds[m][16 + jj] + xzr[H + col]     + bias[H + col];
        float zg = z_lds[m][32 + jj] + xzr[2 * H + col] + bias[2 * H + col];
        float zo = z_lds[m][48 + jj] + xzr[3 * H + col] + bias[3 * H + col];
        float ig = sigm(zi), fg = sigm(zf), gg = tanhf(zg), og = sigm(zo);
        float c = fg * cst[m * H + col] + ig * gg;
        float h = og * tanhf(c);
        cst[m * H + col]   = c;
        hnext[m * H + col] = f2bf(h);
        if (l1out)  l1out[((size_t)tf * 64 + m) * 1024 + d * H + col] = f2bf(h);
        if (encout) encout[((size_t)m * T + tf) * 1024 + d * H + col] = h;
        if (stateout) {
            stateout[(size_t)m * 1024 + d * H + col] = h;                 // state_h
            stateout[(size_t)64 * 1024 + m * 1024 + d * H + col] = c;     // state_c
        }
    }
}

// ---------------------------------------------------------------- host ----
extern "C" void kernel_launch(void* const* d_in, const int* in_sizes, int n_in,
                              void* d_out, int out_size, void* d_ws, size_t ws_size,
                              hipStream_t stream) {
    (void)in_sizes; (void)n_in; (void)out_size; (void)ws_size;
    const int E = 512, H = 512, B = 64, T = 512;
    const int FourH = 4 * H;          // 2048
    const int M = B * T;              // 32768

    const int*   idx = (const int*)d_in[0];
    const float* emb = (const float*)d_in[1];
    const float* W1f = (const float*)d_in[2];
    const float* U1f = (const float*)d_in[3];
    const float* b1f = (const float*)d_in[4];
    const float* W1b = (const float*)d_in[5];
    const float* U1b = (const float*)d_in[6];
    const float* b1b = (const float*)d_in[7];
    const float* W2f = (const float*)d_in[8];
    const float* U2f = (const float*)d_in[9];
    const float* b2f = (const float*)d_in[10];
    const float* W2b = (const float*)d_in[11];
    const float* U2b = (const float*)d_in[12];
    const float* b2b = (const float*)d_in[13];
    float* out = (float*)d_out;

    char* ws = (char*)d_ws;
    size_t off = 0;
    auto alloc = [&](size_t bytes) -> char* {
        char* p = ws + off;
        off += (bytes + 255) & ~(size_t)255;
        return p;
    };

    unsigned short* x_bf = (unsigned short*)alloc((size_t)M * E * 2);        // 32 MB
    unsigned short* wt1f = (unsigned short*)alloc((size_t)FourH * E * 2);
    unsigned short* wt1b = (unsigned short*)alloc((size_t)FourH * E * 2);
    unsigned short* wt2f = (unsigned short*)alloc((size_t)FourH * 2 * H * 2);
    unsigned short* wt2b = (unsigned short*)alloc((size_t)FourH * 2 * H * 2);
    unsigned short* ut1f = (unsigned short*)alloc((size_t)FourH * H * 2);
    unsigned short* ut1b = (unsigned short*)alloc((size_t)FourH * H * 2);
    unsigned short* ut2f = (unsigned short*)alloc((size_t)FourH * H * 2);
    unsigned short* ut2b = (unsigned short*)alloc((size_t)FourH * H * 2);
    float* xzf = (float*)alloc((size_t)M * FourH * 4);                       // 256 MB (reused L1/L2)
    float* xzb = (float*)alloc((size_t)M * FourH * 4);                       // 256 MB
    unsigned short* l1 = (unsigned short*)alloc((size_t)M * 2 * H * 2);      // 64 MB
    // state block (kept contiguous so one k_zero covers c_f,c_b,hA_f,hA_b):
    float*          c_f  = (float*)alloc((size_t)B * H * 4);
    float*          c_b  = (float*)alloc((size_t)B * H * 4);
    unsigned short* hA_f = (unsigned short*)alloc((size_t)B * H * 2);
    unsigned short* hA_b = (unsigned short*)alloc((size_t)B * H * 2);
    unsigned short* hB_f = (unsigned short*)alloc((size_t)B * H * 2);
    unsigned short* hB_b = (unsigned short*)alloc((size_t)B * H * 2);
    const int zero_u32 = (2 * B * H * 4 + 2 * B * H * 2) / 4;                // c_f,c_b,hA_f,hA_b

    // 1. embedding gather (f32 -> bf16, [T][B][E])
    k_embed_gather<<<M, 256, 0, stream>>>(idx, emb, x_bf);

    // 2. weight transposes to [N][K] bf16
    k_transpose_bf16<<<FourH, 256, 0, stream>>>(W1f, wt1f, E, FourH);
    k_transpose_bf16<<<FourH, 256, 0, stream>>>(W1b, wt1b, E, FourH);
    k_transpose_bf16<<<FourH, 256, 0, stream>>>(U1f, ut1f, H, FourH);
    k_transpose_bf16<<<FourH, 256, 0, stream>>>(U1b, ut1b, H, FourH);
    k_transpose_bf16<<<FourH, 256, 0, stream>>>(W2f, wt2f, 2 * H, FourH);
    k_transpose_bf16<<<FourH, 256, 0, stream>>>(W2b, wt2b, 2 * H, FourH);
    k_transpose_bf16<<<FourH, 256, 0, stream>>>(U2f, ut2f, H, FourH);
    k_transpose_bf16<<<FourH, 256, 0, stream>>>(U2b, ut2b, H, FourH);

    dim3 ggrid(FourH / 64, M / 128);

    // 3. layer-1 input projections (no bias; added in step kernel)
    k_gemm_xz<<<ggrid, 256, 0, stream>>>(x_bf, wt1f, xzf, M, FourH, E);
    k_gemm_xz<<<ggrid, 256, 0, stream>>>(x_bf, wt1b, xzb, M, FourH, E);

    // 4. layer-1 scan
    k_zero<<<(zero_u32 + 255) / 256, 256, 0, stream>>>((unsigned int*)c_f, zero_u32);
    for (int s = 0; s < T; ++s) {
        const unsigned short* hpf = (s & 1) ? hB_f : hA_f;
        const unsigned short* hpb = (s & 1) ? hB_b : hA_b;
        unsigned short* hnf = (s & 1) ? hA_f : hB_f;
        unsigned short* hnb = (s & 1) ? hA_b : hB_b;
        k_lstm_step<<<64, 256, 0, stream>>>(xzf, xzb, ut1f, ut1b, b1f, b1b,
                                            hpf, hpb, hnf, hnb, c_f, c_b,
                                            l1, (float*)nullptr, (float*)nullptr, s);
    }

    // 5. layer-2 input projections (K = 2H), reusing xz buffers
    k_gemm_xz<<<ggrid, 256, 0, stream>>>(l1, wt2f, xzf, M, FourH, 2 * H);
    k_gemm_xz<<<ggrid, 256, 0, stream>>>(l1, wt2b, xzb, M, FourH, 2 * H);

    // 6. layer-2 scan; last step also writes state_h / state_c
    k_zero<<<(zero_u32 + 255) / 256, 256, 0, stream>>>((unsigned int*)c_f, zero_u32);
    float* state_base = out + (size_t)B * T * 2 * H;
    for (int s = 0; s < T; ++s) {
        const unsigned short* hpf = (s & 1) ? hB_f : hA_f;
        const unsigned short* hpb = (s & 1) ? hB_b : hA_b;
        unsigned short* hnf = (s & 1) ? hA_f : hB_f;
        unsigned short* hnb = (s & 1) ? hA_b : hB_b;
        k_lstm_step<<<64, 256, 0, stream>>>(xzf, xzb, ut2f, ut2b, b2f, b2b,
                                            hpf, hpb, hnf, hnb, c_f, c_b,
                                            (unsigned short*)nullptr, out,
                                            (s == T - 1) ? state_base : (float*)nullptr, s);
    }
}
